// EdgeAwareGNN_20684562497885
// MI455X (gfx1250) — matrix operand order repaired and checked by reading, not compiled, not run on hardware
//
#include <hip/hip_runtime.h>
#include <hip/hip_bf16.h>

// ---------------------------------------------------------------------------
// EdgeAwareGNN on MI455X (gfx1250, wave32, WMMA).
//
// Dominant cost: per-edge weight MLP GEMM of layer 2:
//   (E x m) @ (m x m), m = 4096  ->  ~840 GFLOP (bf16 WMMA, f32 accum).
// A-matrix relu(e*w1+b1) is generated on the fly into LDS (32 edges x m bf16,
// ~262 KB -- needs CDNA5's 320 KB WGP LDS), B-matrix is w2 transposed +
// converted to bf16 (33.5 MB, L2-resident). Each wave runs TWO 16-row WMMA
// sub-tiles against every B fragment it loads (2x arithmetic intensity vs L2).
// GEMM output is fused directly into msg[e,o] += x[src_e,i] * C[e, i*64+o],
// never materializing the E x m weight tensor. Segment-mean via f32 atomics.
// ---------------------------------------------------------------------------

#define NN 20000
#define EE 25000
#define LN_EPS 1e-5f

typedef __attribute__((ext_vector_type(16))) __bf16 bf16x16;
typedef __attribute__((ext_vector_type(8)))  __bf16 bf16x8;
typedef __attribute__((ext_vector_type(8)))  float  f32x8;

// ---------------------------------------------------------------------------
// Tiled transpose + f32 -> bf16 convert:  w2t[j][k] = bf16(w2[k][j])
// ---------------------------------------------------------------------------
template<int M>
__global__ __launch_bounds__(256)
void transpose_bf16_kernel(const float* __restrict__ w2, __bf16* __restrict__ w2t)
{
    __shared__ float tile[32][33];
    const int k0 = blockIdx.x * 32;
    const int j0 = blockIdx.y * 32;
    const int tx = threadIdx.x, ty = threadIdx.y;          // 32 x 8
    for (int yy = ty; yy < 32; yy += 8)
        tile[yy][tx] = w2[(size_t)(k0 + yy) * M + j0 + tx];
    __syncthreads();
    for (int yy = ty; yy < 32; yy += 8)
        w2t[(size_t)(j0 + yy) * M + k0 + tx] = (__bf16)tile[tx][yy];
}

// ---------------------------------------------------------------------------
// Fused NNConv edge kernel (d_out == 64). One workgroup = 32 edges
// (two 16-row WMMA sub-tiles sharing each B load).
//   r[e][k] = relu(ev[e]*w1[k] + b1[k])          (bf16, in LDS, built once)
//   C[e][j] = sum_k r[e][k] * w2[k][j]           (WMMA bf16 16x16x32)
//   msg[e][o] = sum_i x[src_e][i] * (C[e][i*64+o] + b2[i*64+o])
//   agg[dst_e][o] += msg[e][o] ; cnt[dst_e] += 1
// wave w owns the o-block [16w, 16w+16): its v8f accumulators match the
// WMMA C layout (row = v + 8*(lane>=16), col = lane&15) exactly.
// ---------------------------------------------------------------------------
template<int M, int D_IN>
__global__ __launch_bounds__(128)
void nnconv_edge_kernel(const float* __restrict__ x,
                        const int*   __restrict__ src,
                        const int*   __restrict__ dst,
                        const float* __restrict__ ev,
                        const float* __restrict__ w1,
                        const float* __restrict__ b1,
                        const __bf16* __restrict__ w2t,   // M x M, [j][k]
                        const float* __restrict__ b2,
                        float* __restrict__ agg,          // N x 64
                        float* __restrict__ cnt)          // N
{
    constexpr int TE  = 32;          // edges per workgroup (2 WMMA row-tiles)
    constexpr int RS  = M + 8;       // padded bf16 row stride (16B aligned rows)
    constexpr int NJT = M / 16;      // output-column tiles

    extern __shared__ char smem_raw[];
    __bf16* rbuf = reinterpret_cast<__bf16*>(smem_raw);                        // TE x RS
    float*  xs   = reinterpret_cast<float*>(smem_raw + (size_t)TE * RS * 2);   // TE x D_IN
    float*  msg  = xs + TE * D_IN;                                             // TE x 64
    float*  eL   = msg + TE * 64;                                              // TE
    int*    dL   = reinterpret_cast<int*>(eL + TE);                            // TE

    const int tid  = threadIdx.x;
    const int wave = tid >> 5;
    const int lane = tid & 31;
    const int base = blockIdx.x * TE;

    if (tid < TE) {
        const int e = base + tid;
        const bool v = (e < EE);
        eL[tid] = v ? ev[e]  : 0.0f;
        dL[tid] = v ? dst[e] : 0;
    }
    __syncthreads();

    // gather x rows (zeroed for out-of-range edges -> msg contribution = 0)
    for (int idx = tid; idx < TE * D_IN; idx += 128) {
        const int r = idx / D_IN, c = idx % D_IN;
        const int e = base + r;
        xs[idx] = (e < EE) ? x[(size_t)src[e] * D_IN + c] : 0.0f;
    }
    // build A-tile: r[e][k] = relu(e*w1[k]+b1[k]), bf16
    for (int p = tid; p < TE * M; p += 128) {
        const int r = p / M, k = p % M;
        const float v = fmaxf(fmaf(eL[r], w1[k], b1[k]), 0.0f);
        rbuf[r * RS + k] = (__bf16)v;
    }
    __syncthreads();

    // msg init with the b2 bias term: sum_i x[e,i] * b2[i*64+o]
    for (int idx = tid; idx < TE * 64; idx += 128) {
        const int r = idx >> 6, o = idx & 63;
        float s = 0.0f;
        #pragma unroll
        for (int i = 0; i < D_IN; ++i)
            s = fmaf(xs[r * D_IN + i], b2[i * 64 + o], s);
        msg[idx] = s;
    }
    __syncthreads();

    // --- WMMA GEMM + fused contraction with x ---
    // A layout (16-bit 16x32): lanes 0-15 row m, K 0-7 & 16-23; lanes 16-31 K 8-15 & 24-31
    const int  rowA  = lane & 15;
    const int  koffA = (lane >> 4) << 3;     // 0 or 8
    // B layout (16-bit 32x16): lanes 0-15 col n, K 0-7 & 16-23; lanes 16-31 col n, K 8-15 & 24-31
    const int  colB  = lane & 15;
    const int  koffB = (lane >> 4) << 3;     // 0 or 8
    const int  rb    = (lane >> 4) << 3;     // row base for C layout (0 or 8)

    f32x8 acc0 = {};
    f32x8 acc1 = {};
    const __bf16* ap0 = rbuf + rowA * RS + koffA;           // rows 0-15
    const __bf16* ap1 = ap0 + 16 * RS;                      // rows 16-31

    for (int jt = wave; jt < NJT; jt += 4) {       // each wave: fixed q = wave
        const int jb = jt << 4;
        const __bf16* bp = w2t + (size_t)(jb + colB) * M + koffB;

        f32x8 c0 = {};
        f32x8 c1 = {};
        #pragma unroll 4
        for (int kb = 0; kb < M; kb += 32) {
            union { bf16x16 v; bf16x8 h[2]; } a0, a1, b;
            b.h[0]  = *reinterpret_cast<const bf16x8*>(bp + kb);
            b.h[1]  = *reinterpret_cast<const bf16x8*>(bp + kb + 16);
            a0.h[0] = *reinterpret_cast<const bf16x8*>(ap0 + kb);
            a0.h[1] = *reinterpret_cast<const bf16x8*>(ap0 + kb + 16);
            a1.h[0] = *reinterpret_cast<const bf16x8*>(ap1 + kb);
            a1.h[1] = *reinterpret_cast<const bf16x8*>(ap1 + kb + 16);
            c0 = __builtin_amdgcn_wmma_f32_16x16x32_bf16(
                    false, a0.v, false, b.v, (short)0, c0, false, false);
            c1 = __builtin_amdgcn_wmma_f32_16x16x32_bf16(
                    false, a1.v, false, b.v, (short)0, c1, false, false);
        }
        const int i = jt >> 2;                     // j = i*64 + o, tile shares i
        #pragma unroll
        for (int v = 0; v < 8; ++v) {
            acc0[v] = fmaf(xs[(rb + v) * D_IN + i],      c0[v], acc0[v]);
            acc1[v] = fmaf(xs[(16 + rb + v) * D_IN + i], c1[v], acc1[v]);
        }
    }

    // each (e,o) cell has exactly one (wave,lane,v) writer
    {
        const int o = (wave << 4) + (lane & 15);
        #pragma unroll
        for (int v = 0; v < 8; ++v) {
            msg[(rb + v) * 64 + o]      += acc0[v];
            msg[(16 + rb + v) * 64 + o] += acc1[v];
        }
    }
    __syncthreads();

    // scatter to segment sums
    for (int idx = tid; idx < TE * 64; idx += 128) {
        const int r = idx >> 6;
        if (base + r < EE)
            atomicAdd(&agg[(size_t)dL[r] * 64 + (idx & 63)], msg[idx]);
    }
    if (tid < TE && base + tid < EE)
        atomicAdd(&cnt[dL[tid]], 1.0f);
}

// ---------------------------------------------------------------------------
// mean + bias + LayerNorm + relu, one node per 64-thread block
// ---------------------------------------------------------------------------
__global__ __launch_bounds__(64)
void finalize_ln_kernel(const float* __restrict__ agg, const float* __restrict__ cnt,
                        const float* __restrict__ bias, const float* __restrict__ g,
                        const float* __restrict__ be, float* __restrict__ h)
{
    __shared__ float red[64];
    const int n = blockIdx.x, o = threadIdx.x;
    const float v = agg[(size_t)n * 64 + o] / fmaxf(cnt[n], 1.0f) + bias[o];
    red[o] = v; __syncthreads();
    for (int s = 32; s > 0; s >>= 1) { if (o < s) red[o] += red[o + s]; __syncthreads(); }
    const float mu = red[0] * (1.0f / 64.0f); __syncthreads();
    const float d = v - mu;
    red[o] = d * d; __syncthreads();
    for (int s = 32; s > 0; s >>= 1) { if (o < s) red[o] += red[o + s]; __syncthreads(); }
    const float var = red[0] * (1.0f / 64.0f);
    const float y = d * __frsqrt_rn(var + LN_EPS) * g[o] + be[o];
    h[(size_t)n * 64 + o] = fmaxf(y, 0.0f);
}

// ---------------------------------------------------------------------------
// Layer 3 (d_out = 1):  msg_e = relu(e*w1+b1) . (W2^T x_src) + b2 . x_src
// Per-node precompute V[n,k] = sum_i h2[n,i]*w2[k,i], c[n] = sum_i h2[n,i]*b2[i]
// ---------------------------------------------------------------------------
__global__ __launch_bounds__(64)
void l3_node_kernel(const float* __restrict__ h2, const float* __restrict__ w2,
                    const float* __restrict__ b2, float* __restrict__ V,
                    float* __restrict__ c)
{
    __shared__ float xr[64];
    const int n = blockIdx.x, k = threadIdx.x;
    xr[k] = h2[(size_t)n * 64 + k];
    __syncthreads();
    float s = 0.0f;
    #pragma unroll 8
    for (int i = 0; i < 64; ++i) s = fmaf(xr[i], w2[k * 64 + i], s);
    V[(size_t)n * 64 + k] = s;
    if (k == 0) {
        float cc = 0.0f;
        for (int i = 0; i < 64; ++i) cc = fmaf(xr[i], b2[i], cc);
        c[n] = cc;
    }
}

__global__ __launch_bounds__(256)
void l3_edge_kernel(const int* __restrict__ src, const int* __restrict__ dst,
                    const float* __restrict__ ev, const float* __restrict__ w1,
                    const float* __restrict__ b1, const float* __restrict__ V,
                    const float* __restrict__ c, float* __restrict__ agg3,
                    float* __restrict__ cnt3)
{
    const int e = blockIdx.x * blockDim.x + threadIdx.x;
    if (e >= EE) return;
    const float ee = ev[e];
    const int s = src[e];
    const float* vp = V + (size_t)s * 64;
    float m = c[s];
    #pragma unroll 8
    for (int k = 0; k < 64; ++k)
        m = fmaf(fmaxf(fmaf(ee, w1[k], b1[k]), 0.0f), vp[k], m);
    atomicAdd(&agg3[dst[e]], m);
    atomicAdd(&cnt3[dst[e]], 1.0f);
}

__global__ __launch_bounds__(256)
void out_kernel(const float* __restrict__ agg3, const float* __restrict__ cnt3,
                const float* __restrict__ bias3, float* __restrict__ out)
{
    const int n = blockIdx.x * blockDim.x + threadIdx.x;
    if (n >= NN) return;
    const float v = agg3[n] / fmaxf(cnt3[n], 1.0f) + bias3[0];
    out[n] = (v > 20.0f) ? v : log1pf(__expf(v));   // softplus
}

// ---------------------------------------------------------------------------
extern "C" void kernel_launch(void* const* d_in, const int* in_sizes, int n_in,
                              void* d_out, int out_size, void* d_ws, size_t ws_size,
                              hipStream_t stream)
{
    (void)in_sizes; (void)n_in; (void)out_size; (void)ws_size;

    const float* x     = (const float*)d_in[0];
    const int*   src1  = (const int*)d_in[1];
    const int*   dst1  = (const int*)d_in[2];
    const float* e1    = (const float*)d_in[3];
    const int*   src2  = (const int*)d_in[4];
    const int*   dst2  = (const int*)d_in[5];
    const float* e2    = (const float*)d_in[6];
    const int*   src3  = (const int*)d_in[7];
    const int*   dst3  = (const int*)d_in[8];
    const float* e3    = (const float*)d_in[9];
    const float* ew1_1 = (const float*)d_in[10];
    const float* eb1_1 = (const float*)d_in[11];
    const float* ew2_1 = (const float*)d_in[12];
    const float* eb2_1 = (const float*)d_in[13];
    const float* bias1 = (const float*)d_in[14];
    const float* ew1_2 = (const float*)d_in[15];
    const float* eb1_2 = (const float*)d_in[16];
    const float* ew2_2 = (const float*)d_in[17];
    const float* eb2_2 = (const float*)d_in[18];
    const float* bias2 = (const float*)d_in[19];
    const float* ew1_3 = (const float*)d_in[20];
    const float* eb1_3 = (const float*)d_in[21];
    const float* ew2_3 = (const float*)d_in[22];
    const float* eb2_3 = (const float*)d_in[23];
    const float* bias3 = (const float*)d_in[24];
    const float* g1    = (const float*)d_in[25];
    const float* be1   = (const float*)d_in[26];
    const float* g2    = (const float*)d_in[27];
    const float* be2   = (const float*)d_in[28];

    // ---- workspace layout (accumulators first so one memset zeroes them) ----
    char*  ws  = (char*)d_ws;
    size_t off = 0;
    auto alloc = [&](size_t bytes) -> char* {
        char* p = ws + off;
        off += (bytes + 255) & ~(size_t)255;
        return p;
    };
    float* agg1 = (float*)alloc((size_t)NN * 64 * 4);
    float* agg2 = (float*)alloc((size_t)NN * 64 * 4);
    float* agg3 = (float*)alloc((size_t)NN * 4);
    float* cnt1 = (float*)alloc((size_t)NN * 4);
    float* cnt2 = (float*)alloc((size_t)NN * 4);
    float* cnt3 = (float*)alloc((size_t)NN * 4);
    const size_t zero_bytes = off;
    float*  h1   = (float*)alloc((size_t)NN * 64 * 4);
    float*  h2   = (float*)alloc((size_t)NN * 64 * 4);
    float*  Vbuf = (float*)alloc((size_t)NN * 64 * 4);
    float*  cbuf = (float*)alloc((size_t)NN * 4);
    __bf16* w2t1 = (__bf16*)alloc((size_t)384 * 384 * 2);
    __bf16* w2t2 = (__bf16*)alloc((size_t)4096 * 4096 * 2);

    hipMemsetAsync(d_ws, 0, zero_bytes, stream);

    // ---- B-matrix prep: transpose + bf16 convert (L2-resident afterwards) ----
    {
        dim3 tb(32, 8);
        transpose_bf16_kernel<384> <<<dim3(384 / 32, 384 / 32),   tb, 0, stream>>>(ew2_1, w2t1);
        transpose_bf16_kernel<4096><<<dim3(4096 / 32, 4096 / 32), tb, 0, stream>>>(ew2_2, w2t2);
    }

    const int nblk = (EE + 31) / 32;

    // ---- layer 1: IN=6 -> H=64, m=384 ----
    {
        constexpr int M = 384, DIN = 6;
        const size_t shmem = (size_t)32 * (M + 8) * 2 + (32 * DIN + 32 * 64 + 32) * 4 + 32 * 4;
        nnconv_edge_kernel<M, DIN><<<nblk, 128, shmem, stream>>>(
            x, src1, dst1, e1, ew1_1, eb1_1, w2t1, eb2_1, agg1, cnt1);
        finalize_ln_kernel<<<NN, 64, 0, stream>>>(agg1, cnt1, bias1, g1, be1, h1);
    }
    // ---- layer 2: H=64 -> H=64, m=4096 (the 840 GFLOP GEMM) ----
    {
        constexpr int M = 4096, DIN = 64;
        const size_t shmem = (size_t)32 * (M + 8) * 2 + (32 * DIN + 32 * 64 + 32) * 4 + 32 * 4;
        hipFuncSetAttribute(reinterpret_cast<const void*>(nnconv_edge_kernel<M, DIN>),
                            hipFuncAttributeMaxDynamicSharedMemorySize, (int)shmem);
        nnconv_edge_kernel<M, DIN><<<nblk, 128, shmem, stream>>>(
            h1, src2, dst2, e2, ew1_2, eb1_2, w2t2, eb2_2, agg2, cnt2);
        finalize_ln_kernel<<<NN, 64, 0, stream>>>(agg2, cnt2, bias2, g2, be2, h2);
    }
    // ---- layer 3: H=64 -> 1 (refactored, scalar) ----
    l3_node_kernel<<<NN, 64, 0, stream>>>(h2, ew2_3, eb2_3, Vbuf, cbuf);
    l3_edge_kernel<<<(EE + 255) / 256, 256, 0, stream>>>(
        src3, dst3, e3, ew1_3, eb1_3, Vbuf, cbuf, agg3, cnt3);
    out_kernel<<<(NN + 255) / 256, 256, 0, stream>>>(agg3, cnt3, bias3, (float*)d_out);
}
